// SimpleGraphLayer_86620900426036
// MI455X (gfx1250) — matrix operand order repaired and checked by reading, not compiled, not run on hardware
//
#include <hip/hip_runtime.h>
#include <hip/hip_bf16.h>

// ---------------------------------------------------------------------------
// out = relu(adj @ (x @ W^T) + b)
//
// Associativity folds the 64x64 linear layer into the aggregation GEMM:
// hot loop = [16384 x 16384] @ [16384 x 64]. adj (1 GiB fp32) streams from
// HBM exactly once -> ~46us floor at 23.3 TB/s (AI = 32 FLOP/B; break-even
// matrix rate 745 TFLOP/s, far below the bf16 WMMA ceiling). adj is read
// fp32 (its HBM bytes are irreducible), converted to bf16 in registers,
// accumulated fp32 via v_wmma_f32_16x16x32_bf16.
//
// B operand (y = x@W^T, 2 MiB bf16, pre-packed in exact B-fragment layout)
// is staged per-block through double-buffered LDS via the CDNA5 async path
// (GLOBAL_LOAD_ASYNC_TO_LDS_B128, ASYNCcnt): the per-chunk barrier waits
// only on asynccnt + a raw split barrier, so the next chunk's A loads stay
// in flight across it.
// ---------------------------------------------------------------------------

typedef __bf16 bf16_t;
typedef __attribute__((ext_vector_type(16))) __bf16 v16bf;
typedef __attribute__((ext_vector_type(8)))  float  v8f;
typedef __attribute__((ext_vector_type(4)))  float  v4f;
typedef __attribute__((ext_vector_type(4)))  int    v4i;
typedef __attribute__((address_space(1))) v4i* gv4i_p;   // global v4i*
typedef __attribute__((address_space(3))) v4i* lv4i_p;   // LDS v4i*

#define NN     16384          // nodes (rows/cols of adj)
#define DIM    64             // in_dim == out_dim
#define NCHUNK (NN / 32)      // K-chunks of 32 for 16x16x32 bf16 WMMA
#define CHUNK_ELEMS (4 * 32 * 16)   // bf16 elems per packed chunk (4 KiB)

#if defined(__has_builtin)
#  if __has_builtin(__builtin_amdgcn_global_load_async_to_lds_b128) && \
      __has_builtin(__builtin_amdgcn_s_wait_asynccnt)
#    define ASYNC_STAGE 1
#  else
#    define ASYNC_STAGE 0
#  endif
#  if __has_builtin(__builtin_amdgcn_s_barrier_signal) && \
      __has_builtin(__builtin_amdgcn_s_barrier_wait)
#    define RAW_BARRIER 1
#  else
#    define RAW_BARRIER 0
#  endif
#else
#  define ASYNC_STAGE 0
#  define RAW_BARRIER 0
#endif

// ---------------------------------------------------------------------------
// Kernel 1: y = x @ W^T, stored bf16 pre-packed in WMMA B-fragment layout
// (32x16 tile: lanes 0-15 hold K=0-7,16-23, lanes 16-31 hold K=8-15,24-31,
// two 16-bit K values per VGPR).  Packed index:
//   yp[ ((c*4 + tn)*32 + lane)*16 + i ],  lane = 16*g + (n%16),
//   k = c*32 + (i<8 ? 8*g + i : 16 + 8*g + (i-8)),  n = tn*16 + (n%16).
// ---------------------------------------------------------------------------
__global__ void __launch_bounds__(256)
gcn_pack_y(const float* __restrict__ x, const float* __restrict__ W,
           bf16_t* __restrict__ yp)
{
    int tid = blockIdx.x * 256 + threadIdx.x;   // one thread per (k, n)
    int k = tid >> 6;
    int n = tid & 63;
    const float* xr = x + (size_t)k * DIM;
    const float* wr = W + (size_t)n * DIM;
    float s = 0.f;
#pragma unroll
    for (int j = 0; j < DIM; ++j) s = fmaf(xr[j], wr[j], s);

    int c   = k >> 5;
    int kin = k & 31;
    int g   = (kin >> 3) & 1;
    int i   = (kin & 7) | ((kin >> 4) << 3);
    int tn  = n >> 4;
    int lanep = (g << 4) | (n & 15);
    yp[(size_t)((c * 4 + tn) * 32 + lanep) * 16 + i] = (bf16_t)s;
}

// ---------------------------------------------------------------------------
// Kernel 2: out = relu(adj @ y + b).
// 4 waves/block, wave w owns rows [blockIdx*64 + w*16, +16) x all 64 cols.
// A fragment (16-bit 16x32 layout): lane l -> m = l%16, g = l/16 holds
// K = {8g..8g+7} U {16+8g..16+8g+7}: two contiguous 8-float runs; lanes l
// and l+16 jointly cover a full 128B cacheline of each adj row.
// ---------------------------------------------------------------------------

__device__ __forceinline__ void wg_barrier_only()
{
#if RAW_BARRIER
    asm volatile("" ::: "memory");
    __builtin_amdgcn_s_barrier_signal(-1);
    __builtin_amdgcn_s_barrier_wait(-1);
    asm volatile("" ::: "memory");
#else
    __syncthreads();
#endif
}

__device__ __forceinline__ void stage_chunk(const bf16_t* __restrict__ yp,
                                            bf16_t* lds_dst, int c, int tid)
{
#if ASYNC_STAGE
    // memory -> LDS with no VGPR transit, tracked by ASYNCcnt.
    // Each thread moves 32B = 2 x b128; the immediate offset is applied to
    // both the global and LDS addresses (CDNA5 ISA).
    gv4i_p g = (gv4i_p)(yp + (size_t)c * CHUNK_ELEMS + (size_t)tid * 16);
    lv4i_p l = (lv4i_p)(lds_dst + (size_t)tid * 16);
    __builtin_amdgcn_global_load_async_to_lds_b128(g, l, 0,  0);
    __builtin_amdgcn_global_load_async_to_lds_b128(g, l, 16, 0);
#else
    ((v16bf*)lds_dst)[tid] =
        ((const v16bf*)yp)[(size_t)c * 128 + tid];
#endif
}

__device__ __forceinline__ void stage_fence()
{
#if ASYNC_STAGE
    __builtin_amdgcn_s_wait_asynccnt(0);   // our LDS writes are committed
    wg_barrier_only();                     // everyone's writes visible
#else
    __syncthreads();
#endif
}

__device__ __forceinline__ void chunk_wmma(const v4f a0, const v4f a1,
                                           const v4f a2, const v4f a3,
                                           const v16bf* __restrict__ fb,
                                           int lane, v8f acc[4])
{
    v16bf af;
    af[0]  = (bf16_t)a0[0]; af[1]  = (bf16_t)a0[1];
    af[2]  = (bf16_t)a0[2]; af[3]  = (bf16_t)a0[3];
    af[4]  = (bf16_t)a1[0]; af[5]  = (bf16_t)a1[1];
    af[6]  = (bf16_t)a1[2]; af[7]  = (bf16_t)a1[3];
    af[8]  = (bf16_t)a2[0]; af[9]  = (bf16_t)a2[1];
    af[10] = (bf16_t)a2[2]; af[11] = (bf16_t)a2[3];
    af[12] = (bf16_t)a3[0]; af[13] = (bf16_t)a3[1];
    af[14] = (bf16_t)a3[2]; af[15] = (bf16_t)a3[3];
#pragma unroll
    for (int tn = 0; tn < 4; ++tn) {
        v16bf bfrag = fb[tn * 32 + lane];        // ds_load_b128 x2
        acc[tn] = __builtin_amdgcn_wmma_f32_16x16x32_bf16(
            false, af, false, bfrag, (short)0, acc[tn], false, false);
    }
}

__global__ void __launch_bounds__(128)
gcn_aggregate(const float* __restrict__ adj, const bf16_t* __restrict__ yp,
              const float* __restrict__ bias, float* __restrict__ out)
{
    __shared__ __align__(16) bf16_t ybuf[2][CHUNK_ELEMS];   // 2 x 4 KiB

    const int tid  = threadIdx.x;
    const int lane = tid & 31;
    const int wave = tid >> 5;
    const int rbase = blockIdx.x * 64 + wave * 16;
    const int m = lane & 15;
    const int g = lane >> 4;

    const float* pa = adj + (size_t)(rbase + m) * NN + g * 8;

    // ---- prologue: stage chunk 0 into LDS, load chunk 0 A into regs ----
    stage_chunk(yp, &ybuf[0][0], 0, tid);
    v4f a0 = *(const v4f*)(pa);
    v4f a1 = *(const v4f*)(pa + 4);
    v4f a2 = *(const v4f*)(pa + 16);
    v4f a3 = *(const v4f*)(pa + 20);
    pa += 32;
    stage_fence();

    v8f acc[4] = {v8f{}, v8f{}, v8f{}, v8f{}};

    // ---- steady state: fill chunk c+1 (regs + other LDS buffer) while
    //      computing chunk c; barrier waits only on staging (asynccnt),
    //      never on the in-flight A loads ------------------------------
    for (int c = 0; c < NCHUNK - 1; ++c) {
        const int buf = c & 1;
        stage_chunk(yp, &ybuf[buf ^ 1][0], c + 1, tid);
        // next A tile
        v4f n0 = *(const v4f*)(pa);
        v4f n1 = *(const v4f*)(pa + 4);
        v4f n2 = *(const v4f*)(pa + 16);
        v4f n3 = *(const v4f*)(pa + 20);
        __builtin_prefetch(pa + 512, 0, 3);      // warm L2 ~2KB ahead
        pa += 32;

        chunk_wmma(a0, a1, a2, a3, (const v16bf*)&ybuf[buf][0], lane, acc);

        stage_fence();
        a0 = n0; a1 = n1; a2 = n2; a3 = n3;
    }

    // ---- drain: last chunk -------------------------------------------
    chunk_wmma(a0, a1, a2, a3,
               (const v16bf*)&ybuf[(NCHUNK - 1) & 1][0], lane, acc);

    // ---- epilogue: bias + relu. D layout: lane l -> n = l%16,
    //      rows m = r + 8*(l/16) for VGPR r = 0..7 ----------------------
    const int ncol = lane & 15;
    float bn[4];
#pragma unroll
    for (int tn = 0; tn < 4; ++tn) bn[tn] = bias[tn * 16 + ncol];

    const int mrow0 = rbase + (lane >> 4) * 8;
#pragma unroll
    for (int r = 0; r < 8; ++r) {
        float* orow = out + (size_t)(mrow0 + r) * DIM + ncol;
#pragma unroll
        for (int tn = 0; tn < 4; ++tn) {
            float v = acc[tn][r] + bn[tn];
            orow[tn * 16] = v > 0.f ? v : 0.f;
        }
    }
}

// ---------------------------------------------------------------------------
extern "C" void kernel_launch(void* const* d_in, const int* in_sizes, int n_in,
                              void* d_out, int out_size, void* d_ws, size_t ws_size,
                              hipStream_t stream)
{
    const float* x    = (const float*)d_in[0];   // [N, 64]
    const float* adj  = (const float*)d_in[1];   // [N, N]
    const float* W    = (const float*)d_in[2];   // [64, 64]
    const float* bias = (const float*)d_in[3];   // [64]
    float*       out  = (float*)d_out;           // [N, 64]
    bf16_t*      yp   = (bf16_t*)d_ws;           // 2 MiB packed B operand

    // Pre-pass: y = x @ W^T, packed into WMMA-B fragment layout.
    gcn_pack_y<<<(NN * DIM) / 256, 256, 0, stream>>>(x, W, yp);

    // Hot loop: out = relu(adj @ y + b), streaming adj once at HBM rate.
    gcn_aggregate<<<NN / 64, 128, 0, stream>>>(adj, yp, bias, out);
}
